// pred_net_45457933860878
// MI455X (gfx1250) — compile-verified, matrix operand order checked by reading, not run on hardware
//
#include <hip/hip_runtime.h>
#include <hip/hip_bf16.h>
#include <stdint.h>

// ---------------------------------------------------------------------------
// Types for CDNA5 WMMA (wave32): bf16 A/B fragments, f32 accumulator.
// ---------------------------------------------------------------------------
typedef uint16_t u16;
typedef u16    u16x8   __attribute__((ext_vector_type(8)));
typedef __bf16 bf16x16 __attribute__((ext_vector_type(16)));
typedef float  f32x8   __attribute__((ext_vector_type(8)));

union FragU { bf16x16 v; u16x8 h[2]; };

__device__ __forceinline__ u16 f2bf(float f) {
    union { float f; uint32_t u; } c; c.f = f;
    uint32_t u = c.u;
    u += 0x7FFFu + ((u >> 16) & 1u);   // round-to-nearest-even
    return (u16)(u >> 16);
}

// ---------------------------------------------------------------------------
// Weight conversion kernels (run once per launch; weights then live in L2)
// ---------------------------------------------------------------------------
__global__ void k_f32_to_bf16(const float* __restrict__ in, u16* __restrict__ out, int n) {
    int i = blockIdx.x * blockDim.x + threadIdx.x;
    int stride = gridDim.x * blockDim.x;
    for (; i < n; i += stride) out[i] = f2bf(in[i]);
}

// fc1eff[n,k] = fc1[n,k] + fc1[n,2048+k]   (combined = [h,h] folding)
__global__ void k_fc1_fold(const float* __restrict__ fc1, u16* __restrict__ out) {
    int i = blockIdx.x * blockDim.x + threadIdx.x;
    int stride = gridDim.x * blockDim.x;
    const int NK = 2048 * 2048;
    for (; i < NK; i += stride) {
        int nrow = i >> 11, k = i & 2047;
        out[i] = f2bf(fc1[nrow * 4096 + k] + fc1[nrow * 4096 + 2048 + k]);
    }
}

// ---------------------------------------------------------------------------
// Activation pack buffer Apack: 256 x 2240 bf16 = [h | act_t | latent]
// ---------------------------------------------------------------------------
__global__ void k_pack_init(const float* __restrict__ state, const float* __restrict__ act,
                            const float* __restrict__ latent, u16* __restrict__ Apack) {
    int i = blockIdx.x * blockDim.x + threadIdx.x;
    int stride = gridDim.x * blockDim.x;
    const int TOT = 256 * 2240;
    for (; i < TOT; i += stride) {
        int b = i / 2240, c = i - b * 2240;
        float v;
        if (c < 2048)       v = state[b * 2048 + c];
        else if (c < 2112)  v = act[(b * 32 + 0) * 64 + (c - 2048)];   // act[:,0,:]
        else                v = latent[b * 128 + (c - 2112)];
        Apack[i] = f2bf(v);
    }
}

// ---------------------------------------------------------------------------
// bf16 WMMA GEMM: C[MxN] = act( A[MxK] * Bw[NxK]^T + bias ), M = 256
// WG tile 64(M) x 128(N), 128 threads = 4 waves (2x2), wave tile 32x64.
// Per K-chunk (32): 8 WMMAs vs 12 ds_load_b128 fragment reads per wave.
// ---------------------------------------------------------------------------
#define TM 64
#define TN 128
#define TK 32
#define LDS_STRIDE 40   // 32 + 8 pad -> 80B rows, conflict-free ds_load_b128

__global__ __launch_bounds__(128) void k_gemm(
    const u16* __restrict__ A, int lda,
    const u16* __restrict__ Bw, int ldb,
    int K,
    const float* __restrict__ bias,
    u16* __restrict__ outB,
    float* __restrict__ outF,
    int ldc, int do_relu)
{
    __shared__ alignas(16) u16 sA[TM * LDS_STRIDE];
    __shared__ alignas(16) u16 sB[TN * LDS_STRIDE];

    const int tid  = threadIdx.x;
    const int lane = tid & 31;
    const int wave = tid >> 5;    // 0..3
    const int wm   = wave & 1;    // two 32-row wave blocks
    const int wn   = wave >> 1;   // two 64-col wave blocks
    const int r    = lane & 15;
    const int half = lane >> 4;

    const int m0 = blockIdx.y * TM;
    const int n0 = blockIdx.x * TN;

    f32x8 acc[2][4];
    #pragma unroll
    for (int i = 0; i < 2; i++)
        #pragma unroll
        for (int j = 0; j < 4; j++)
            #pragma unroll
            for (int e = 0; e < 8; e++) acc[i][j][e] = 0.0f;

    for (int kc = 0; kc < K; kc += TK) {
        // A tile: 64x32 = 256 16B-vectors / 128 thr -> 2 each
        #pragma unroll
        for (int s = 0; s < 2; s++) {
            const int g = tid + 128 * s;
            const int row = g >> 2, col = (g & 3) * 8;
            *(u16x8*)&sA[row * LDS_STRIDE + col] =
                *(const u16x8*)&A[(size_t)(m0 + row) * lda + kc + col];
        }
        // B tile: 128x32 = 512 vectors -> 4 each
        #pragma unroll
        for (int s = 0; s < 4; s++) {
            const int g = tid + 128 * s;
            const int row = g >> 2, col = (g & 3) * 8;
            *(u16x8*)&sB[row * LDS_STRIDE + col] =
                *(const u16x8*)&Bw[(size_t)(n0 + row) * ldb + kc + col];
        }
        __syncthreads();

        FragU af[2], bf[4];
        #pragma unroll
        for (int mi = 0; mi < 2; mi++) {
            const u16* p = &sA[(32 * wm + 16 * mi + r) * LDS_STRIDE + 8 * half];
            af[mi].h[0] = *(const u16x8*)p;          // K = kc + 8*half + 0..7
            af[mi].h[1] = *(const u16x8*)(p + 16);   // K = kc + 16 + 8*half + 0..7
        }
        #pragma unroll
        for (int ni = 0; ni < 4; ni++) {
            const u16* p = &sB[(64 * wn + 16 * ni + r) * LDS_STRIDE + 8 * half];
            bf[ni].h[0] = *(const u16x8*)p;
            bf[ni].h[1] = *(const u16x8*)(p + 16);
        }
        #pragma unroll
        for (int mi = 0; mi < 2; mi++)
            #pragma unroll
            for (int ni = 0; ni < 4; ni++)
                acc[mi][ni] = __builtin_amdgcn_wmma_f32_16x16x32_bf16(
                    false, af[mi].v, false, bf[ni].v,
                    (short)0, acc[mi][ni], false, false);
        __syncthreads();
    }

    // Epilogue: C element e -> m = e + 8*half, n = lane%16
    #pragma unroll
    for (int mi = 0; mi < 2; mi++) {
        #pragma unroll
        for (int ni = 0; ni < 4; ni++) {
            const int ng = n0 + 64 * wn + 16 * ni + r;
            const float bv = bias ? bias[ng] : 0.0f;
            #pragma unroll
            for (int e = 0; e < 8; e++) {
                const int mg = m0 + 32 * wm + 16 * mi + 8 * half + e;
                float v = acc[mi][ni][e] + bv;
                if (do_relu) v = fmaxf(v, 0.0f);
                if (outF) outF[(size_t)mg * ldc + ng] = v;
                if (outB) outB[(size_t)mg * ldc + ng] = f2bf(v);
            }
        }
    }
}

// ---------------------------------------------------------------------------
// LSTM gate math: consumes cc (256x8192 fp32, already ReLU'd), updates c,
// writes preds[t] fp32, h_next bf16 into Apack[:, :2048], and (fused) the
// act_t columns of Apack for the carry GEMM.
// ---------------------------------------------------------------------------
__global__ void k_gates(const float* __restrict__ cc, float* __restrict__ c,
                        float* __restrict__ preds, u16* __restrict__ Apack,
                        const float* __restrict__ act, int t) {
    int i = blockIdx.x * blockDim.x + threadIdx.x;   // 256*2048
    if (i >= 256 * 2048) return;
    int b = i >> 11, j = i & 2047;
    const float* row = cc + (size_t)b * 8192;
    float ig = 1.0f / (1.0f + __expf(-row[j]));
    float fg = 1.0f / (1.0f + __expf(-row[j + 2048]));
    float og = 1.0f / (1.0f + __expf(-row[j + 4096]));
    float gg = tanhf(row[j + 6144]);
    float cn = fg * c[i] + ig * gg;
    float hn = og * tanhf(cn);
    c[i] = cn;
    preds[((size_t)t * 256 + b) * 2048 + j] = hn;
    Apack[b * 2240 + j] = f2bf(hn);
    if (j < 64)  // fused act_t repack: Apack[:, 2048:2112] = act[:, t, :]
        Apack[b * 2240 + 2048 + j] = f2bf(act[(b * 32 + t) * 64 + j]);
}

// reward[t*B+b] = dot(h_next[b], reward_W) + reward_b
__global__ void k_reward(const float* __restrict__ preds, const float* __restrict__ rw,
                         const float* __restrict__ rb, float* __restrict__ rewards, int t) {
    __shared__ float red[256];
    const int b = blockIdx.x;
    const float* h = preds + ((size_t)t * 256 + b) * 2048;
    float s = 0.0f;
    for (int k = threadIdx.x; k < 2048; k += 256) s += h[k] * rw[k];
    red[threadIdx.x] = s;
    __syncthreads();
    for (int off = 128; off > 0; off >>= 1) {
        if (threadIdx.x < off) red[threadIdx.x] += red[threadIdx.x + off];
        __syncthreads();
    }
    if (threadIdx.x == 0) rewards[t * 256 + b] = red[0] + rb[0];
}

// ---------------------------------------------------------------------------
// Host: sequence of launches (graph-capture safe, deterministic).
// ---------------------------------------------------------------------------
extern "C" void kernel_launch(void* const* d_in, const int* in_sizes, int n_in,
                              void* d_out, int out_size, void* d_ws, size_t ws_size,
                              hipStream_t stream) {
    (void)in_sizes; (void)n_in; (void)out_size; (void)ws_size;
    const float* state  = (const float*)d_in[0];
    const float* act    = (const float*)d_in[1];
    const float* latent = (const float*)d_in[2];
    const float* fc_W   = (const float*)d_in[3];
    const float* fc_b   = (const float*)d_in[4];
    const float* fc1_W  = (const float*)d_in[5];
    const float* fc2_W  = (const float*)d_in[6];
    const float* W_W    = (const float*)d_in[7];
    const float* rw     = (const float*)d_in[8];
    const float* rb     = (const float*)d_in[9];

    float* preds   = (float*)d_out;                         // [32*256, 2048]
    float* rewards = preds + (size_t)32 * 256 * 2048;       // [32*256, 1]

    char* ws = (char*)d_ws;
    size_t off = 0;
    auto alloc = [&](size_t bytes) -> void* {
        void* p = ws + off;
        off = (off + bytes + 255) & ~(size_t)255;
        return p;
    };
    u16*   wfc   = (u16*)  alloc((size_t)2048 * 2240 * 2);  // fc_W bf16 (N x K)
    u16*   wfc1  = (u16*)  alloc((size_t)2048 * 2048 * 2);  // folded fc1 bf16
    u16*   wfc2  = (u16*)  alloc((size_t)2048 * 2048 * 2);
    u16*   wW    = (u16*)  alloc((size_t)8192 * 2048 * 2);
    u16*   Apack = (u16*)  alloc((size_t)256 * 2240 * 2);
    u16*   hbuf  = (u16*)  alloc((size_t)256 * 2048 * 2);   // carry h, bf16
    u16*   x1    = (u16*)  alloc((size_t)256 * 2048 * 2);
    u16*   x2    = (u16*)  alloc((size_t)256 * 2048 * 2);
    float* cc    = (float*)alloc((size_t)256 * 8192 * 4);
    float* cbuf  = (float*)alloc((size_t)256 * 2048 * 4);   // cell state fp32

    // one-time (per launch) weight prep; bf16 weights total 60MB -> stay in L2
    k_f32_to_bf16<<<1024, 256, 0, stream>>>(fc_W,  wfc,  2048 * 2240);
    k_fc1_fold   <<<1024, 256, 0, stream>>>(fc1_W, wfc1);
    k_f32_to_bf16<<<1024, 256, 0, stream>>>(fc2_W, wfc2, 2048 * 2048);
    k_f32_to_bf16<<<2048, 256, 0, stream>>>(W_W,   wW,   8192 * 2048);
    k_pack_init  <<<1024, 256, 0, stream>>>(state, act, latent, Apack);

    const dim3 blk(128);
    // h0 = relu(fc([state, act0, latent])); cell = h0 (fp32 copy)
    k_gemm<<<dim3(16, 4), blk, 0, stream>>>(Apack, 2240, wfc, 2240, 2240,
                                            fc_b, hbuf, cbuf, 2048, 1);

    for (int t = 0; t < 32; t++) {
        // x1 = relu(h @ fc1eff^T)
        k_gemm<<<dim3(16, 4), blk, 0, stream>>>(hbuf, 2048, wfc1, 2048, 2048,
                                                nullptr, x1, nullptr, 2048, 1);
        // x2 = relu(x1 @ fc2^T)
        k_gemm<<<dim3(16, 4), blk, 0, stream>>>(x1, 2048, wfc2, 2048, 2048,
                                                nullptr, x2, nullptr, 2048, 1);
        // cc = relu(x2 @ W^T)   (256 x 8192, fp32 for gate math)
        k_gemm<<<dim3(64, 4), blk, 0, stream>>>(x2, 2048, wW, 2048, 2048,
                                                nullptr, nullptr, cc, 8192, 1);
        // gates -> c, h_next; preds[t]; h_next bf16 + act_t -> Apack (fused)
        k_gates<<<2048, 256, 0, stream>>>(cc, cbuf, preds, Apack, act, t);
        // reward[t]
        k_reward<<<256, 256, 0, stream>>>(preds, rw, rb, rewards, t);
        if (t < 31) {
            // carry h = relu(fc([h_next, act_t, latent]))
            k_gemm<<<dim3(16, 4), blk, 0, stream>>>(Apack, 2240, wfc, 2240, 2240,
                                                    fc_b, hbuf, nullptr, 2048, 1);
        }
    }
}
